// CyclicalLangevinSampler_57561151701249
// MI455X (gfx1250) — matrix operand order repaired and checked by reading, not compiled, not run on hardware
//
#include <hip/hip_runtime.h>

// ---------------------------------------------------------------------------
// Discrete cyclical-Langevin MH step, fused for MI455X (gfx1250, wave32).
//   8192 chains x 784 dims.  Two f16 WMMA GEMMs (x@Ws, x_delta@Ws) fused with
//   proposal sampling, MH correction and accept.  Ws = W + W^T is symmetric,
//   so both A and B WMMA fragments are contiguous row loads.
//   Block = 128 threads (4 waves) sharing one 16-row LDS slab; the 49 N-tiles
//   are split across the 4 waves -> 2048 waves total to fill all SIMD32s.
// ---------------------------------------------------------------------------

#define DDIM    784
#define KP      800          // K padded to a multiple of 32
#define NT      49           // 784 / 16 column tiles
#define KT      25           // 800 / 32 k tiles
#define NCHAINS 8192
#define NWAVES  4            // waves per block

typedef __attribute__((ext_vector_type(16))) _Float16     v16h;
typedef __attribute__((ext_vector_type(8)))  float        v8f;
typedef __attribute__((ext_vector_type(4)))  unsigned int u32x4;

union Frag { v16h v; u32x4 q[2]; };

// A-matrix 16x32 f16 fragment: lane ll holds row M=ll; lanes 0-15 carry
// K = k0..k0+7 and k0+16..k0+23, lanes 16-31 carry K = k0+8.. and k0+24..
// -> two 16B chunks at (k0 + hi*8) and (k0 + hi*8 + 16).
__device__ __forceinline__ v16h load_a_frag(const _Float16* p) {
  Frag f;
  f.q[0] = *(const u32x4*)(p);
  f.q[1] = *(const u32x4*)(p + 16);
  return f.v;
}

// Fragment in this code's contiguous k order (A and B): lane L holds row/col L&15, K = k0 + hi*16 + t
// (t = 0..15 contiguous) -> one contiguous 32B chunk.
__device__ __forceinline__ v16h load_b_frag(const _Float16* p) {
  Frag f;
  f.q[0] = *(const u32x4*)(p);
  f.q[1] = *(const u32x4*)(p + 8);
  return f.v;
}

__device__ __forceinline__ float log_sigmoid(float z) {
  // stable: min(z,0) - log1p(exp(-|z|))
  return fminf(z, 0.0f) - __logf(1.0f + __expf(-fabsf(z)));
}

// ---------------------------------------------------------------------------
// Kernel 1: Ws = f16(W + W^T), row-major [784][800], cols 784..799 zero pad.
// ---------------------------------------------------------------------------
__global__ void prep_ws_kernel(const float* __restrict__ W,
                               _Float16* __restrict__ Ws) {
  const int idx = blockIdx.x * blockDim.x + threadIdx.x;
  if (idx >= DDIM * KP) return;
  const int r = idx / KP;
  const int c = idx - r * KP;
  float v = 0.0f;
  if (c < DDIM) v = W[r * DDIM + c] + W[c * DDIM + r];
  Ws[idx] = (_Float16)v;
}

// ---------------------------------------------------------------------------
// Kernel 2: block of 4 waves owns 16 chains; N-tiles split across waves.
// ---------------------------------------------------------------------------
__global__ __launch_bounds__(32 * NWAVES)
void dlmc_step_kernel(const float* __restrict__ x,
                      const float* __restrict__ bvec,
                      const float* __restrict__ rr,
                      const float* __restrict__ rr_acc,
                      const int*   __restrict__ k_iter,
                      const _Float16* __restrict__ Ws,
                      float* __restrict__ out) {
  __shared__ __align__(16) _Float16 sx[16 * KP];   // x rows, f16, K-padded
  __shared__ __align__(16) _Float16 sxd[16 * KP];  // x_delta rows
  __shared__ float s_lpf[16], s_ex[16], s_lpr[16], s_exd[16];
  __shared__ float s_sel[16];                      // per-row accept flag

  const int tid  = threadIdx.x;
  const int wid  = tid >> 5;                       // wave id 0..3
  const int lane = tid & 31;
  const int ll   = lane & 15;
  const int hi   = lane >> 4;
  const int row0 = blockIdx.x * 16;

  // cyclical step-size schedule: step = 0.2*(cos(pi*k/100)+1); term2 = 1/(2*step)
  const float kf    = (float)(*k_iter);
  const float step  = 0.2f * (cosf(3.14159265358979323f * kf / 100.0f) + 1.0f);
  const float term2 = 1.0f / (2.0f * step);

  // Stage x (f32 global -> f16 LDS, zero pad K tail); pre-zero sxd pad;
  // zero the cross-wave reduction scratch.
  for (int idx = tid; idx < 16 * KP; idx += 32 * NWAVES) {
    const int r = idx / KP;
    const int c = idx - r * KP;
    float v = 0.0f;
    if (c < DDIM) v = x[(size_t)(row0 + r) * DDIM + c];
    sx[idx] = (_Float16)v;
    if (c >= DDIM) sxd[idx] = (_Float16)0.0f;
  }
  if (tid < 16) {
    s_lpf[tid] = 0.f; s_ex[tid] = 0.f; s_lpr[tid] = 0.f; s_exd[tid] = 0.f;
  }
  __syncthreads();

  float lpf[8], ex[8], lpr[8], exd[8];
#pragma unroll
  for (int r = 0; r < 8; ++r) { lpf[r] = 0.f; ex[r] = 0.f; lpr[r] = 0.f; exd[r] = 0.f; }

  // ------------- Pass A: gx = x @ Ws (+b), propose flips (wave's N-tiles) --
  for (int nt = wid; nt < NT; nt += NWAVES) {
    const int n0 = nt * 16;
    const _Float16* wrow = Ws + (size_t)(n0 + ll) * KP + hi * 16;
    v8f acc = {};
#pragma unroll 5
    for (int kt = 0; kt < KT; ++kt) {
      v16h af = load_b_frag(sx + ll * KP + kt * 32 + hi * 16);
      v16h bf = load_b_frag(wrow + kt * 32);
      if (kt + 1 < KT) __builtin_prefetch(wrow + (kt + 1) * 32, 0, 1);
      acc = __builtin_amdgcn_wmma_f32_16x16x32_f16(
          false, af, false, bf, (short)0, acc, false, false);
    }
    const int   j  = n0 + ll;                  // global column for this lane
    const float bj = bvec[j];
#pragma unroll
    for (int r = 0; r < 8; ++r) {
      const int   m  = r + hi * 8;             // local row (C layout: M = r + hi*8)
      const float g  = acc[r] + bj;            // gx_full = x@Ws + b
      const float xv = (float)sx[m * KP + j];
      const float zf = -(2.0f * xv - 1.0f) * g - term2;
      const float flip = 1.0f / (1.0f + __expf(-zf));
      const float u  = rr[(size_t)(row0 + m) * DDIM + j];
      const bool  ind = u < flip;
      const float xd = ind ? (1.0f - xv) : xv;
      lpf[r] += log_sigmoid(ind ? zf : -zf);
      ex[r]  += xv * (g + bj);                 // 2*E(x) partial: x·gx_full + x·b
      sxd[m * KP + j] = (_Float16)xd;
    }
  }
  __syncthreads();                             // all x_delta columns visible

  // ------------- Pass B: gxd = x_delta @ Ws (+b), reverse terms -----------
  for (int nt = wid; nt < NT; nt += NWAVES) {
    const int n0 = nt * 16;
    const _Float16* wrow = Ws + (size_t)(n0 + ll) * KP + hi * 16;
    v8f acc = {};
#pragma unroll 5
    for (int kt = 0; kt < KT; ++kt) {
      v16h af = load_b_frag(sxd + ll * KP + kt * 32 + hi * 16);
      v16h bf = load_b_frag(wrow + kt * 32);
      if (kt + 1 < KT) __builtin_prefetch(wrow + (kt + 1) * 32, 0, 1);
      acc = __builtin_amdgcn_wmma_f32_16x16x32_f16(
          false, af, false, bf, (short)0, acc, false, false);
    }
    const int   j  = n0 + ll;
    const float bj = bvec[j];
#pragma unroll
    for (int r = 0; r < 8; ++r) {
      const int   m   = r + hi * 8;
      const float g2  = acc[r] + bj;           // gxd_full
      const float xdv = (float)sxd[m * KP + j];
      const float xv  = (float)sx[m * KP + j];
      const bool  ind = (xdv != xv);
      const float zr  = -(2.0f * xdv - 1.0f) * g2 - term2;
      lpr[r] += log_sigmoid(ind ? zr : -zr);
      exd[r] += xdv * (g2 + bj);               // 2*E(x_delta) partial
    }
  }

  // In-wave butterfly reduce across the 16 lanes of each C-matrix half
  // (xor masks 1,2,4,8 stay within lanes 0-15 / 16-31).
#pragma unroll
  for (int r = 0; r < 8; ++r) {
#pragma unroll
    for (int msk = 1; msk <= 8; msk <<= 1) {
      lpf[r] += __shfl_xor(lpf[r], msk, 32);
      ex[r]  += __shfl_xor(ex[r],  msk, 32);
      lpr[r] += __shfl_xor(lpr[r], msk, 32);
      exd[r] += __shfl_xor(exd[r], msk, 32);
    }
  }

  // Merge the 4 waves' partial sums in LDS (ds_add_f32 atomics).
  if (ll == 0) {                               // lanes 0 (rows 0-7) and 16 (rows 8-15)
#pragma unroll
    for (int r = 0; r < 8; ++r) {
      const int m = r + hi * 8;
      atomicAdd(&s_lpf[m], lpf[r]);
      atomicAdd(&s_ex[m],  ex[r]);
      atomicAdd(&s_lpr[m], lpr[r]);
      atomicAdd(&s_exd[m], exd[r]);
    }
  }
  __syncthreads();

  // Accept test: la = (E(xd)-E(x)) + lp_reverse - lp_forward, per row.
  if (tid < 16) {
    const int   m  = tid;
    const float la = 0.5f * (s_exd[m] - s_ex[m]) + s_lpr[m] - s_lpf[m];
    const float ua = rr_acc[row0 + m];
    s_sel[m] = (__expf(la) > ua) ? 1.0f : 0.0f;
  }
  __syncthreads();

  // Coalesced f32 writeout: x_new = accept ? x_delta : x
  for (int idx = tid; idx < 16 * DDIM; idx += 32 * NWAVES) {
    const int m = idx / DDIM;
    const int c = idx - m * DDIM;
    const float xv = (float)sx[m * KP + c];
    const float xd = (float)sxd[m * KP + c];
    out[(size_t)(row0 + m) * DDIM + c] = (s_sel[m] > 0.5f) ? xd : xv;
  }
}

// ---------------------------------------------------------------------------
extern "C" void kernel_launch(void* const* d_in, const int* in_sizes, int n_in,
                              void* d_out, int out_size, void* d_ws, size_t ws_size,
                              hipStream_t stream) {
  const float* x      = (const float*)d_in[0];
  const float* W      = (const float*)d_in[1];
  const float* b      = (const float*)d_in[2];
  const float* rr     = (const float*)d_in[3];
  const float* rr_acc = (const float*)d_in[4];
  const int*   k_iter = (const int*)d_in[5];
  float*       out    = (float*)d_out;
  _Float16*    Ws     = (_Float16*)d_ws;     // 784*800*2 B = 1.25 MB scratch

  (void)in_sizes; (void)n_in; (void)out_size; (void)ws_size;

  const int n1 = DDIM * KP;
  prep_ws_kernel<<<(n1 + 255) / 256, 256, 0, stream>>>(W, Ws);
  dlmc_step_kernel<<<NCHAINS / 16, 32 * NWAVES, 0, stream>>>(
      x, b, rr, rr_acc, k_iter, Ws, out);
}
